// MAB_16123307229385
// MI455X (gfx1250) — matrix operand order repaired and checked by reading, not compiled, not run on hardware
//
#include <hip/hip_runtime.h>
#include <hip/hip_bf16.h>

#define DIM   512
#define BATCH 32
#define SQL   512
#define SKL   512
#define NH    8
#define HD    64

typedef __attribute__((ext_vector_type(16))) __bf16 v16bf;
typedef __attribute__((ext_vector_type(8)))  float  v8f;

// -------- helpers --------------------------------------------------------

// Native RNE f32->bf16 convert (backend emits v_cvt_pk_bf16_f32 pairs).
__device__ __forceinline__ __bf16 f2bf(float f) { return (__bf16)f; }

__device__ __forceinline__ void storeOT(float* p, float v)  { *p = v; }
__device__ __forceinline__ void storeOT(__bf16* p, float v) { *p = f2bf(v); }

// A-fragment (16x32 bf16): this lane holds K = base+0..15 of its row
// (contiguous); ap already points at row + kk + base.
__device__ __forceinline__ v16bf load_a_frag(const float* __restrict__ ap) {
  const float4 a0 = *(const float4*)(ap + 0);
  const float4 a1 = *(const float4*)(ap + 4);
  const float4 a2 = *(const float4*)(ap + 8);
  const float4 a3 = *(const float4*)(ap + 12);
  v16bf a;
  a[0]  = f2bf(a0.x); a[1]  = f2bf(a0.y); a[2]  = f2bf(a0.z); a[3]  = f2bf(a0.w);
  a[4]  = f2bf(a1.x); a[5]  = f2bf(a1.y); a[6]  = f2bf(a1.z); a[7]  = f2bf(a1.w);
  a[8]  = f2bf(a2.x); a[9]  = f2bf(a2.y); a[10] = f2bf(a2.z); a[11] = f2bf(a2.w);
  a[12] = f2bf(a3.x); a[13] = f2bf(a3.y); a[14] = f2bf(a3.z); a[15] = f2bf(a3.w);
  return a;
}

// B-fragment (32x16 bf16): this lane holds 16 contiguous K values of one column.
__device__ __forceinline__ v16bf load_b_frag(const float* __restrict__ bp) {
  const float4 b0 = *(const float4*)(bp + 0);
  const float4 b1 = *(const float4*)(bp + 4);
  const float4 b2 = *(const float4*)(bp + 8);
  const float4 b3 = *(const float4*)(bp + 12);
  v16bf b;
  b[0]  = f2bf(b0.x); b[1]  = f2bf(b0.y); b[2]  = f2bf(b0.z); b[3]  = f2bf(b0.w);
  b[4]  = f2bf(b1.x); b[5]  = f2bf(b1.y); b[6]  = f2bf(b1.z); b[7]  = f2bf(b1.w);
  b[8]  = f2bf(b2.x); b[9]  = f2bf(b2.y); b[10] = f2bf(b2.z); b[11] = f2bf(b2.w);
  b[12] = f2bf(b3.x); b[13] = f2bf(b3.y); b[14] = f2bf(b3.z); b[15] = f2bf(b3.w);
  return b;
}

// -------- GEMM: Out[M,512] = X[M,512] @ W[512,512]^T + bias  -------------
// One wave computes a 16x64 tile (one A fragment reused over 4 N-subtiles).
// RELU_RES: Out = X + relu(acc + bias)   (N == K == 512 residual path)

template <typename OT, bool RELU_RES>
__global__ void __launch_bounds__(32)
gemm16x64(const float* __restrict__ X, const float* __restrict__ W,
          const float* __restrict__ bias, OT* __restrict__ Out) {
  const int lane  = threadIdx.x & 31;
  const int r     = lane & 15;
  const int hi    = lane >> 4;
  const int abase = hi * 16;      // A-fragment K base for this half-wave
  const int bbase = hi * 16;      // B-fragment K base for this half-wave
  const int m0 = blockIdx.y << 4;
  const int n0 = blockIdx.x << 6;

  v8f acc[4] = {};
  const float* xrow = X + (size_t)(m0 + r) * DIM;

  for (int kk = 0; kk < DIM; kk += 32) {
    const v16bf a = load_a_frag(xrow + kk + abase);
    const float* w0 = W + (size_t)(n0 + r) * DIM + kk + bbase;
#pragma unroll
    for (int t = 0; t < 4; ++t) {
      const v16bf b = load_b_frag(w0 + (size_t)t * 16 * DIM);
      acc[t] = __builtin_amdgcn_wmma_f32_16x16x32_bf16(
          false, a, false, b, (short)0, acc[t], false, false);
    }
  }

#pragma unroll
  for (int t = 0; t < 4; ++t) {
    const int col  = n0 + t * 16 + r;
    const float bc = bias[col];
#pragma unroll
    for (int i = 0; i < 8; ++i) {
      const int row = m0 + i + hi * 8;
      float v = acc[t][i] + bc;
      if (RELU_RES) v = fmaxf(v, 0.0f) + X[(size_t)row * DIM + col];
      storeOT(Out + (size_t)row * DIM + col, v);
    }
  }
}

// -------- Attention: scores -> softmax -> AV, residual on projected Q ----
// One wave per (batch, head, 16-row q tile). 16x512 fp32 score strip in LDS.
// V head-slab is DMA'd global->LDS with async-to-LDS (ASYNCcnt), double
// buffered in 128-row chunks; first chunk overlaps QK^T + softmax.

__global__ void __launch_bounds__(32)
attn_kernel(const float* __restrict__ qb, const __bf16* __restrict__ kb,
            const __bf16* __restrict__ vb, float* __restrict__ out) {
  __shared__ float  sc[16][SKL];          // 32 KB score strip
  __shared__ __bf16 vs[2][128][HD];       // 2 x 16 KB V double buffer

  const int lane  = threadIdx.x & 31;
  const int r     = lane & 15;
  const int hi    = lane >> 4;
  const int abase = hi * 16;
  const int bbase = hi * 16;
  const int q0 = blockIdx.x << 4;
  const int h  = blockIdx.y;
  const int b  = blockIdx.z;

  // ---- async stage of one 128-row V chunk (128x64 bf16 = 16 KB) -------
  // 1024 x 16B chunks / 32 lanes = 32 async b128 ops per stage.
  const __bf16* vhead = vb + (size_t)b * SKL * DIM + h * HD;
  auto stage_v = [&](int buf, int k0) {
    const unsigned lbase = (unsigned)(size_t)(&vs[buf][0][0]);
#pragma unroll 4
    for (int j = 0; j < 32; ++j) {
      const int idx  = j * 32 + lane;     // 0..1023
      const int krow = idx >> 3;          // 8 x 16B chunks per 128B row
      const int cc   = idx & 7;
      const unsigned long long ga =
          (unsigned long long)(const void*)(vhead + (size_t)(k0 + krow) * DIM + cc * 8);
      const unsigned la = lbase + (unsigned)((krow * HD + cc * 8) * 2);
      asm volatile("global_load_async_to_lds_b128 %0, %1, off"
                   :: "v"(la), "v"(ga) : "memory");
    }
  };

  stage_v(0, 0);   // overlaps with QK^T + softmax below

  // ---- phase 1: S = q . k^T / sqrt(512) -------------------------------
  const float* qrow = qb + ((size_t)b * SQL + q0 + r) * DIM + h * HD;
  v16bf aq[2];
#pragma unroll
  for (int kkI = 0; kkI < 2; ++kkI)
    aq[kkI] = load_a_frag(qrow + kkI * 32 + abase);

  const float scale = 0.044194173824159216f;          // 1/sqrt(512)
  for (int kt = 0; kt < SKL / 16; ++kt) {
    v8f s = {};
#pragma unroll
    for (int kkI = 0; kkI < 2; ++kkI) {
      // B column = key position kt*16 + r; 16 contiguous bf16 along head dim
      const __bf16* kp = kb + ((size_t)b * SKL + kt * 16 + r) * DIM +
                         h * HD + kkI * 32 + bbase;
      const v16bf bk = *(const v16bf*)kp;
      s = __builtin_amdgcn_wmma_f32_16x16x32_bf16(
          false, aq[kkI], false, bk, (short)0, s, false, false);
    }
#pragma unroll
    for (int i = 0; i < 8; ++i)
      sc[i + hi * 8][kt * 16 + r] = s[i] * scale;
  }
  __syncthreads();

  // ---- phase 2: softmax (lane owns half a row, merge with shfl_xor 16) -
  {
    float m = -3.402823466e+38f;
    const int c0 = hi * 256;
    for (int c = 0; c < 256; ++c) m = fmaxf(m, sc[r][c0 + c]);
    m = fmaxf(m, __shfl_xor(m, 16, 32));
    float sum = 0.0f;
    for (int c = 0; c < 256; ++c) {
      const float p = __expf(sc[r][c0 + c] - m);
      sc[r][c0 + c] = p;
      sum += p;
    }
    sum += __shfl_xor(sum, 16, 32);
    const float inv = 1.0f / sum;
    for (int c = 0; c < 256; ++c) sc[r][c0 + c] *= inv;
  }
  __syncthreads();

  // ---- phase 3: O = A @ V  (16x64 per head), V chunks from LDS --------
  v8f o[4] = {};
  for (int c = 0; c < 4; ++c) {
    if (c < 3) {
      stage_v((c + 1) & 1, (c + 1) * 128);   // prefetch next chunk
      // ASYNCcnt decrements in order: <=32 outstanding => chunk c landed.
      asm volatile("s_wait_asynccnt 0x20" ::: "memory");
    } else {
      asm volatile("s_wait_asynccnt 0x0" ::: "memory");
    }
    const __bf16 (*vbuf)[HD] = vs[c & 1];
    const int kglob0 = c * 128;
    for (int ks = 0; ks < 128; ks += 32) {
      v16bf ap;
#pragma unroll
      for (int e = 0; e < 8; ++e) {
        ap[e]     = f2bf(sc[r][kglob0 + ks + abase + e]);
        ap[e + 8] = f2bf(sc[r][kglob0 + ks + abase + 8 + e]);
      }
#pragma unroll
      for (int t = 0; t < 4; ++t) {
        v16bf bv;
#pragma unroll
        for (int i = 0; i < 16; ++i)
          bv[i] = vbuf[ks + bbase + i][t * 16 + r];
        o[t] = __builtin_amdgcn_wmma_f32_16x16x32_bf16(
            false, ap, false, bv, (short)0, o[t], false, false);
      }
    }
  }

  // ---- epilogue: residual on projected q ------------------------------
#pragma unroll
  for (int t = 0; t < 4; ++t) {
    const int col = h * HD + t * 16 + r;
#pragma unroll
    for (int i = 0; i < 8; ++i) {
      const int row = q0 + i + hi * 8;
      const size_t idx = ((size_t)b * SQL + row) * DIM + col;
      out[idx] = qb[idx] + o[t][i];
    }
  }
}

// -------- LayerNorm over last dim (512), one row per 256-thread block ---

__global__ void __launch_bounds__(256)
ln_kernel(const float* __restrict__ x, const float* __restrict__ g,
          const float* __restrict__ bt, float* __restrict__ y) {
  __shared__ float ssum[8], ssq[8], stot[2];
  const size_t row = blockIdx.x;
  const int t = threadIdx.x;
  const float* xr = x + row * DIM;
  const float v0 = xr[t], v1 = xr[t + 256];
  float s = v0 + v1, q = v0 * v0 + v1 * v1;
#pragma unroll
  for (int o = 16; o > 0; o >>= 1) {
    s += __shfl_xor(s, o, 32);
    q += __shfl_xor(q, o, 32);
  }
  if ((t & 31) == 0) { ssum[t >> 5] = s; ssq[t >> 5] = q; }
  __syncthreads();
  if (t == 0) {
    float S = 0.0f, Q = 0.0f;
#pragma unroll
    for (int i = 0; i < 8; ++i) { S += ssum[i]; Q += ssq[i]; }
    stot[0] = S * (1.0f / DIM);
    stot[1] = Q * (1.0f / DIM);
  }
  __syncthreads();
  const float mu = stot[0];
  const float rs = rsqrtf(stot[1] - mu * mu + 1e-5f);
  y[row * DIM + t]       = (v0 - mu) * rs * g[t]       + bt[t];
  y[row * DIM + t + 256] = (v1 - mu) * rs * g[t + 256] + bt[t + 256];
}

// -------- launch ---------------------------------------------------------

extern "C" void kernel_launch(void* const* d_in, const int* in_sizes, int n_in,
                              void* d_out, int out_size, void* d_ws, size_t ws_size,
                              hipStream_t stream) {
  const float* Kin = (const float*)d_in[0];
  const float* Vin = (const float*)d_in[1];
  const float* Qin = (const float*)d_in[2];
  const float* Wq  = (const float*)d_in[3];
  const float* bq  = (const float*)d_in[4];
  const float* Wk  = (const float*)d_in[5];
  const float* bk  = (const float*)d_in[6];
  const float* Wv  = (const float*)d_in[7];
  const float* bv  = (const float*)d_in[8];
  const float* Wo  = (const float*)d_in[9];
  const float* bo  = (const float*)d_in[10];
  const float* g0  = (const float*)d_in[11];
  const float* b0  = (const float*)d_in[12];
  const float* g1  = (const float*)d_in[13];
  const float* b1  = (const float*)d_in[14];
  float* out = (float*)d_out;

  char* ws = (char*)d_ws;
  const size_t QB = (size_t)BATCH * SQL * DIM * sizeof(float);          // 33.5 MB
  const size_t KB = (size_t)BATCH * SKL * DIM * sizeof(unsigned short); // 16.8 MB
  float*  qb = (float*)ws;               // q projection (fp32, later reused as LN0)
  __bf16* kb = (__bf16*)(ws + QB);       // k projection (bf16)
  __bf16* vb = (__bf16*)(ws + QB + KB);  // v projection (bf16)
  float*  mo = (float*)(ws + QB);        // MLP output (reuses k|v region)

  const dim3 gg(DIM / 64, (BATCH * SQL) / 16);   // (8, 1024)

  // 1-3: input projections (q fp32, k/v bf16)
  gemm16x64<float,  false><<<gg, 32, 0, stream>>>(Qin, Wq, bq, qb);
  gemm16x64<__bf16, false><<<gg, 32, 0, stream>>>(Kin, Wk, bk, kb);
  gemm16x64<__bf16, false><<<gg, 32, 0, stream>>>(Vin, Wv, bv, vb);

  // 4: attention + residual-on-projected-q -> stage through d_out
  attn_kernel<<<dim3(SQL / 16, NH, BATCH), 32, 0, stream>>>(qb, kb, vb, out);

  // 5: LN0 -> reuse q buffer
  ln_kernel<<<BATCH * SQL, 256, 0, stream>>>(out, g0, b0, qb);

  // 6: O = LN0 + relu(LN0 @ Wo^T + bo)
  gemm16x64<float, true><<<gg, 32, 0, stream>>>(qb, Wo, bo, mo);

  // 7: LN1 -> final output
  ln_kernel<<<BATCH * SQL, 256, 0, stream>>>(mo, g1, b1, out);
}
